// Decoder2l_9363028705593
// MI455X (gfx1250) — matrix operand order repaired and checked by reading, not compiled, not run on hardware
//
#include <hip/hip_runtime.h>
#include <stdint.h>

// ---------------------------------------------------------------------------
// Types for CDNA5 WMMA (wave32)
// ---------------------------------------------------------------------------
typedef __attribute__((ext_vector_type(16))) __bf16 v16bf;
typedef __attribute__((ext_vector_type(8)))  float  v8f;

union BF16x16 {
    unsigned short u[16];
    unsigned int   w[8];
    uint4          q[2];
    v16bf          v;
};

static __device__ __forceinline__ unsigned short f2b(float f) {
    uint32_t u = __builtin_bit_cast(uint32_t, f);
    uint32_t r = u + 0x7FFFu + ((u >> 16) & 1u);   // round-to-nearest-even
    return (unsigned short)(r >> 16);
}
static __device__ __forceinline__ unsigned int pk2b(float lo, float hi) {
    return (unsigned int)f2b(lo) | ((unsigned int)f2b(hi) << 16);
}
static __device__ __forceinline__ float b2f(unsigned short s) {
    uint32_t u = ((uint32_t)s) << 16;
    return __builtin_bit_cast(float, u);
}
static __device__ __forceinline__ float sigm(float x) {
    return 1.0f / (1.0f + expf(-x));
}

// ---------------------------------------------------------------------------
// WMMA GEMM:  C[M,N] (+)= A[M,K] * W^T, W stored row-major (N,K) bf16.
// A is bf16 (ABF=1) or f32 (converted in-register).
// One wave computes a 16x64 C strip: 1 A-fragment amortized over 4 B-frags /
// 4 WMMAs per K-step. Overhang N-columns are handled by redirecting the B row
// pointer to a zero-filled buffer BEFORE the loop (no in-loop masking), so the
// K-loop is pure loads + 4 back-to-back WMMAs.
//
// Fragment layouts (CDNA5 wave32, 16-bit):
//   A 16x32: lane covers row (lane&15); K = two contiguous runs of 8:
//            [k0 + half*8, +8) and [k0 + 16 + half*8, +8)   (half = lane>=16)
//   B 32x16: lane covers col (lane&15); K = same two runs of 8 as A
//   C 16x16 f32: VGPR r -> m = r + half*8, col = lane&15
// ---------------------------------------------------------------------------
template<bool ABF, bool ACC, bool TANH, bool OUTBF>
__global__ __launch_bounds__(128) void gemm_wmma(
    const void* __restrict__ Aptr, int lda,
    const unsigned short* __restrict__ Bw,
    const unsigned short* __restrict__ zbuf,   // >= K zeroed bf16 values
    void* __restrict__ Cptr, int ldc,
    const float* __restrict__ bias,
    int M, int N, int K)
{
    const int lane    = threadIdx.x & 31;
    const int wave    = threadIdx.x >> 5;
    const int tile_n0 = blockIdx.x * 64;
    const int tile_m  = (blockIdx.y * 4 + wave) * 16;
    if (tile_m >= M || tile_n0 >= N) return;

    const int half = lane >> 4;
    const int l15  = lane & 15;
    const int am   = tile_m + l15;

    // Per-subtile column info; invalid columns read from the zero buffer.
    int  cnv[4], cnc[4];
    bool nv[4];
    const unsigned short* Brow[4];
#pragma unroll
    for (int s = 0; s < 4; ++s) {
        cnv[s]  = tile_n0 + s * 16 + l15;
        nv[s]   = (cnv[s] < N);
        cnc[s]  = nv[s] ? cnv[s] : (N - 1);
        Brow[s] = nv[s] ? (Bw + (size_t)cnv[s] * K) : zbuf;
    }

    const float*          ArowF = (const float*)Aptr + (size_t)am * lda;
    const unsigned short* ArowB = (const unsigned short*)Aptr + (size_t)am * lda;
    const int koffA = half * 8;
    const int koffB = half * 8;

    // Init / preload accumulators
    v8f c[4];
#pragma unroll
    for (int s = 0; s < 4; ++s) {
        float bv = (bias != nullptr) ? bias[cnc[s]] : 0.0f;
#pragma unroll
        for (int r = 0; r < 8; ++r) {
            int m = tile_m + r + half * 8;
            float v = bv;
            if (ACC) v += ((const float*)Cptr)[(size_t)m * ldc + cnc[s]];
            c[s][r] = v;
        }
    }

    for (int k0 = 0; k0 < K; k0 += 32) {
        // ---- A fragment ----
        BF16x16 a;
        if (ABF) {
            a.q[0] = *(const uint4*)(ArowB + k0 + koffA);
            a.q[1] = *(const uint4*)(ArowB + k0 + 16 + koffA);
        } else {
            const float4* a0 = (const float4*)(ArowF + k0 + koffA);
            const float4* a1 = (const float4*)(ArowF + k0 + 16 + koffA);
            float4 f0 = a0[0], f1 = a0[1];
            float4 f2 = a1[0], f3 = a1[1];
            a.w[0] = pk2b(f0.x, f0.y); a.w[1] = pk2b(f0.z, f0.w);
            a.w[2] = pk2b(f1.x, f1.y); a.w[3] = pk2b(f1.z, f1.w);
            a.w[4] = pk2b(f2.x, f2.y); a.w[5] = pk2b(f2.z, f2.w);
            a.w[6] = pk2b(f3.x, f3.y); a.w[7] = pk2b(f3.z, f3.w);
        }

        // ---- 4 B fragments: plain b128 loads, no masking ----
        BF16x16 b0, b1, b2, b3;
        {
            const uint4* q0 = (const uint4*)(Brow[0] + k0 + koffB);
            const uint4* q1 = (const uint4*)(Brow[1] + k0 + koffB);
            const uint4* q2 = (const uint4*)(Brow[2] + k0 + koffB);
            const uint4* q3 = (const uint4*)(Brow[3] + k0 + koffB);
            b0.q[0] = q0[0]; b0.q[1] = q0[2];
            b1.q[0] = q1[0]; b1.q[1] = q1[2];
            b2.q[0] = q2[0]; b2.q[1] = q2[2];
            b3.q[0] = q3[0]; b3.q[1] = q3[2];
        }

        c[0] = __builtin_amdgcn_wmma_f32_16x16x32_bf16(false, a.v, false, b0.v,
                                                       (short)0, c[0], false, false);
        c[1] = __builtin_amdgcn_wmma_f32_16x16x32_bf16(false, a.v, false, b1.v,
                                                       (short)0, c[1], false, false);
        c[2] = __builtin_amdgcn_wmma_f32_16x16x32_bf16(false, a.v, false, b2.v,
                                                       (short)0, c[2], false, false);
        c[3] = __builtin_amdgcn_wmma_f32_16x16x32_bf16(false, a.v, false, b3.v,
                                                       (short)0, c[3], false, false);
    }

    // Store 4 subtiles
#pragma unroll
    for (int s = 0; s < 4; ++s) {
        if (!nv[s]) continue;
#pragma unroll
        for (int r = 0; r < 8; ++r) {
            int m   = tile_m + r + half * 8;
            float v = c[s][r];
            if (TANH) v = tanhf(v);
            if (OUTBF)
                ((unsigned short*)Cptr)[(size_t)m * ldc + cnv[s]] = f2b(v);
            else
                ((float*)Cptr)[(size_t)m * ldc + cnv[s]] = v;
        }
    }
}

// ---------------------------------------------------------------------------
// Elementwise helpers
// ---------------------------------------------------------------------------
__global__ void cvt_f32_bf16(const float* __restrict__ s, unsigned short* __restrict__ d, int n) {
    for (int i = blockIdx.x * blockDim.x + threadIdx.x; i < n; i += gridDim.x * blockDim.x)
        d[i] = f2b(s[i]);
}

__global__ void zero_f32(float* __restrict__ p, int n) {
    for (int i = blockIdx.x * blockDim.x + threadIdx.x; i < n; i += gridDim.x * blockDim.x)
        p[i] = 0.0f;
}

// ---------------------------------------------------------------------------
// Attention conv: convf[b,t,j] = sum_k alpha[b, t+k-50] * Wc[j,0,k]
// ---------------------------------------------------------------------------
__global__ void conv_kernel(const float* __restrict__ alpha, const float* __restrict__ Wc,
                            float* __restrict__ convf, int T, int BT)
{
    int idx = blockIdx.x * blockDim.x + threadIdx.x;
    if (idx >= BT) return;
    int b = idx / T, t = idx % T;
    const float* ab = alpha + (size_t)b * T;
    float acc[10] = {0,0,0,0,0,0,0,0,0,0};
    for (int k = 0; k < 100; ++k) {
        int tt = t + k - 50;
        float a = (tt >= 0 && tt < T) ? ab[tt] : 0.0f;
#pragma unroll
        for (int j = 0; j < 10; ++j) acc[j] += a * Wc[j * 100 + k];
    }
#pragma unroll
    for (int j = 0; j < 10; ++j) convf[(size_t)idx * 10 + j] = acc[j];
}

// ---------------------------------------------------------------------------
// e[b,t] = W_ee . tanh(sproj[b,:] + he[b,t,:] + convf[b,t,:] @ W_fe^T)
// One wave per (b,t); each lane owns 8 contiguous d per iteration (b128 loads
// of the bf16 he stream -- dominant per-step HBM traffic together with gcalc).
// ---------------------------------------------------------------------------
__global__ void attn_e_kernel(const float* __restrict__ sproj,
                              const unsigned short* __restrict__ he_bf,
                              const float* __restrict__ convf,
                              const float* __restrict__ W_fe,
                              const float* __restrict__ W_ee,
                              float* __restrict__ e, int BT, int T, int H2)
{
    int w    = blockIdx.x * (blockDim.x >> 5) + (threadIdx.x >> 5);
    int lane = threadIdx.x & 31;
    if (w >= BT) return;
    int b = w / T;
    const unsigned short* heb = he_bf + (size_t)w * H2;
    const float*          sp  = sproj + (size_t)b * H2;
    float cf[10];
#pragma unroll
    for (int j = 0; j < 10; ++j) cf[j] = convf[(size_t)w * 10 + j];

    float acc = 0.0f;
    for (int d0 = lane * 8; d0 < H2; d0 += 256) {
        uint4 hq = *(const uint4*)(heb + d0);           // 8 bf16
        const unsigned short* hu = (const unsigned short*)&hq;
        float4 s0 = *(const float4*)(sp + d0);
        float4 s1 = *(const float4*)(sp + d0 + 4);
        float4 w0 = *(const float4*)(W_ee + d0);
        float4 w1 = *(const float4*)(W_ee + d0 + 4);
        float sv[8] = {s0.x, s0.y, s0.z, s0.w, s1.x, s1.y, s1.z, s1.w};
        float wv[8] = {w0.x, w0.y, w0.z, w0.w, w1.x, w1.y, w1.z, w1.w};
#pragma unroll
        for (int i = 0; i < 8; ++i) {
            int d = d0 + i;
            float f = 0.0f;
#pragma unroll
            for (int j = 0; j < 10; ++j) f += cf[j] * W_fe[d * 10 + j];
            float v = sv[i] + b2f(hu[i]) + f;
            acc += tanhf(v) * wv[i];
        }
    }
#pragma unroll
    for (int m = 16; m >= 1; m >>= 1) acc += __shfl_xor(acc, m, 32);
    if (lane == 0) e[w] = acc;
}

// ---------------------------------------------------------------------------
// Masked softmax over T (max over full row, per reference). One block per b.
// ---------------------------------------------------------------------------
__global__ void softmax_kernel(const float* __restrict__ e, const int* __restrict__ lengths,
                               float* __restrict__ alpha, int T)
{
    __shared__ float red[256];
    int b = blockIdx.x, tid = threadIdx.x;
    const float* eb = e + (size_t)b * T;

    float mx = -1e30f;
    for (int t = tid; t < T; t += 256) mx = fmaxf(mx, eb[t]);
    red[tid] = mx; __syncthreads();
    for (int s = 128; s > 0; s >>= 1) { if (tid < s) red[tid] = fmaxf(red[tid], red[tid + s]); __syncthreads(); }
    mx = red[0]; __syncthreads();

    int len = lengths[b];
    float sum = 0.0f;
    for (int t = tid; t < T; t += 256) sum += (t < len) ? expf(eb[t] - mx) : 0.0f;
    red[tid] = sum; __syncthreads();
    for (int s = 128; s > 0; s >>= 1) { if (tid < s) red[tid] += red[tid + s]; __syncthreads(); }
    float inv = 1.0f / red[0];

    float* ab = alpha + (size_t)b * T;
    for (int t = tid; t < T; t += 256)
        ab[t] = (t < len) ? expf(eb[t] - mx) * inv : 0.0f;
}

// ---------------------------------------------------------------------------
// g[b,d] = sum_t alpha[b,t] * enc[b,t,d]; reads the bf16 copy of enc (halves
// the per-step stream). Thread owns 8 contiguous d (one b128 per t).
// ---------------------------------------------------------------------------
__global__ void gcalc_kernel(const float* __restrict__ alpha,
                             const unsigned short* __restrict__ enc_bf,
                             float* __restrict__ g, int T, int H2)
{
    int b  = blockIdx.x;
    int d0 = threadIdx.x * 8;                    // 128 threads * 8 = H2
    const float* ab = alpha + (size_t)b * T;
    const unsigned short* eb = enc_bf + (size_t)b * T * H2 + d0;
    float acc[8] = {0,0,0,0,0,0,0,0};
    for (int t = 0; t < T; ++t) {
        float a  = ab[t];
        uint4 e8 = *(const uint4*)(eb + (size_t)t * H2);
        const unsigned short* eu = (const unsigned short*)&e8;
#pragma unroll
        for (int i = 0; i < 8; ++i) acc[i] += a * b2f(eu[i]);
    }
    float* gp = g + (size_t)b * H2 + d0;
#pragma unroll
    for (int i = 0; i < 8; ++i) gp[i] = acc[i];
}

// ---------------------------------------------------------------------------
// r1[b,:] = E_ys[target[b,l], :] + bias[:]
// ---------------------------------------------------------------------------
__global__ void r1init_kernel(const int* __restrict__ target, const float* __restrict__ E_ys,
                              const float* __restrict__ bias, float* __restrict__ r1,
                              int l, int L, int H4, int total)
{
    int idx = blockIdx.x * blockDim.x + threadIdx.x;
    if (idx >= total) return;
    int b = idx / H4, n = idx % H4;
    int tok = target[b * L + l];
    r1[idx] = E_ys[(size_t)tok * H4 + n] + bias[n];
}

// ---------------------------------------------------------------------------
// LSTM cell: r = [i | f | gc | o] (B,4H); updates s,c (B,H)
// ---------------------------------------------------------------------------
__global__ void lstm_kernel(const float* __restrict__ r, float* __restrict__ s,
                            float* __restrict__ c, int H, int total)
{
    int idx = blockIdx.x * blockDim.x + threadIdx.x;
    if (idx >= total) return;
    int b = idx / H, h = idx % H;
    const float* rb = r + (size_t)b * 4 * H;
    float ig = sigm(rb[h]);
    float fg = sigm(rb[H + h]);
    float gc = tanhf(rb[2 * H + h]);
    float og = sigm(rb[3 * H + h]);
    float cn = fg * c[idx] + ig * gc;
    c[idx] = cn;
    s[idx] = og * tanhf(cn);
}

// ---------------------------------------------------------------------------
// Host side
// ---------------------------------------------------------------------------
template<bool ABF, bool ACC, bool TANH, bool OUTBF>
static void launch_gemm(const void* A, int lda, const unsigned short* Bw,
                        const unsigned short* zbuf,
                        void* C, int ldc, const float* bias,
                        int M, int N, int K, hipStream_t s)
{
    int mtiles = (M + 15) / 16;
    dim3 grid((N + 63) / 64, (mtiles + 3) / 4);
    gemm_wmma<ABF, ACC, TANH, OUTBF><<<grid, dim3(128), 0, s>>>(A, lda, Bw, zbuf, C, ldc,
                                                               bias, M, N, K);
}

extern "C" void kernel_launch(void* const* d_in, const int* in_sizes, int n_in,
                              void* d_out, int out_size, void* d_ws, size_t ws_size,
                              hipStream_t stream)
{
    const int B = 32, T = 1000, L = 150, H = 512, V = 5000;
    const int H2 = 2 * H, H4 = 4 * H;
    const int BT = B * T;

    const float* enc     = (const float*)d_in[0];
    const int*   lengths = (const int*)  d_in[1];
    const int*   target  = (const int*)  d_in[2];
    const float* W_se    = (const float*)d_in[3];
    const float* W_he    = (const float*)d_in[4];
    const float* b_he    = (const float*)d_in[5];
    const float* W_ee    = (const float*)d_in[6];
    const float* W_conv  = (const float*)d_in[7];
    const float* W_fe    = (const float*)d_in[8];
    const float* W_sy    = (const float*)d_in[9];
    const float* W_gy    = (const float*)d_in[10];
    const float* b_gy    = (const float*)d_in[11];
    const float* W_yy    = (const float*)d_in[12];
    const float* b_yy    = (const float*)d_in[13];
    const float* E_ys    = (const float*)d_in[14];
    const float* W_ss1   = (const float*)d_in[15];
    const float* W_gs1   = (const float*)d_in[16];
    const float* b_gs1   = (const float*)d_in[17];
    const float* W_ss12  = (const float*)d_in[18];
    const float* W_ss2   = (const float*)d_in[19];
    const float* W_gs2   = (const float*)d_in[20];
    const float* b_gs2   = (const float*)d_in[21];
    float* out = (float*)d_out;

    // Workspace carve-up
    char* ws = (char*)d_ws;
    size_t off = 0;
    auto alloc = [&](size_t bytes) -> char* {
        char* p = ws + off;
        off = (off + bytes + 255) & ~(size_t)255;
        return p;
    };
    unsigned short* enc_bf   = (unsigned short*)alloc((size_t)BT * H2 * 2);
    unsigned short* he_bf    = (unsigned short*)alloc((size_t)BT * H2 * 2);
    unsigned short* Wse_bf   = (unsigned short*)alloc((size_t)H2 * H * 2);
    unsigned short* Whe_bf   = (unsigned short*)alloc((size_t)H2 * H2 * 2);
    unsigned short* Wsy_bf   = (unsigned short*)alloc((size_t)H * H * 2);
    unsigned short* Wgy_bf   = (unsigned short*)alloc((size_t)H * H2 * 2);
    unsigned short* Wyy_bf   = (unsigned short*)alloc((size_t)V * H * 2);
    unsigned short* Wss1_bf  = (unsigned short*)alloc((size_t)H4 * H * 2);
    unsigned short* Wgs1_bf  = (unsigned short*)alloc((size_t)H4 * H2 * 2);
    unsigned short* Wss12_bf = (unsigned short*)alloc((size_t)H4 * H * 2);
    unsigned short* Wss2_bf  = (unsigned short*)alloc((size_t)H4 * H * 2);
    unsigned short* Wgs2_bf  = (unsigned short*)alloc((size_t)H4 * H2 * 2);
    // zero buffer (>= max K bf16 elements) + state block, zeroed every call
    unsigned short* zbuf = (unsigned short*)alloc((size_t)H2 * 2);
    int state_n = BT + 4 * B * H;   // alpha + s1 + c1 + s2 + c2
    float* stateblk = (float*)alloc((size_t)state_n * 4);
    float* alpha = stateblk;
    float* s1 = alpha + BT;
    float* c1 = s1 + B * H;
    float* s2 = c1 + B * H;
    float* c2 = s2 + B * H;
    float* ebuf  = (float*)alloc((size_t)BT * 4);
    float* convf = (float*)alloc((size_t)BT * 10 * 4);
    float* sproj = (float*)alloc((size_t)B * H2 * 4);
    float* g     = (float*)alloc((size_t)B * H2 * 4);
    float* hy    = (float*)alloc((size_t)B * H * 4);
    float* r1    = (float*)alloc((size_t)B * H4 * 4);
    float* r2    = (float*)alloc((size_t)B * H4 * 4);
    (void)ws_size; (void)in_sizes; (void)n_in; (void)out_size;

    // One-time (per call): conversions to bf16 + zeroing + he precompute GEMM
    auto cvt = [&](const float* s, unsigned short* d, int n) {
        int blocks = (n + 255) / 256; if (blocks > 4096) blocks = 4096;
        cvt_f32_bf16<<<blocks, 256, 0, stream>>>(s, d, n);
    };
    cvt(enc,    enc_bf,   BT * H2);
    cvt(W_se,   Wse_bf,   H2 * H);
    cvt(W_he,   Whe_bf,   H2 * H2);
    cvt(W_sy,   Wsy_bf,   H * H);
    cvt(W_gy,   Wgy_bf,   H * H2);
    cvt(W_yy,   Wyy_bf,   V * H);
    cvt(W_ss1,  Wss1_bf,  H4 * H);
    cvt(W_gs1,  Wgs1_bf,  H4 * H2);
    cvt(W_ss12, Wss12_bf, H4 * H);
    cvt(W_ss2,  Wss2_bf,  H4 * H);
    cvt(W_gs2,  Wgs2_bf,  H4 * H2);

    zero_f32<<<(state_n + 255) / 256, 256, 0, stream>>>(stateblk, state_n);
    zero_f32<<<2, 256, 0, stream>>>((float*)zbuf, H2 / 2);

    // he[b,t,:] = enc[b,t,:] @ W_he^T + b_he  (bf16 A, bf16 out; M=BT, N=K=2H)
    launch_gemm<true, false, false, true>(enc_bf, H2, Whe_bf, zbuf, he_bf, H2, b_he,
                                          BT, H2, H2, stream);

    // Sequential decode: 150 steps
    for (int l = 0; l < L; ++l) {
        conv_kernel<<<(BT + 255) / 256, 256, 0, stream>>>(alpha, W_conv, convf, T, BT);
        launch_gemm<false, false, false, false>(s1, H, Wse_bf, zbuf, sproj, H2, nullptr,
                                                B, H2, H, stream);
        attn_e_kernel<<<(BT + 7) / 8, 256, 0, stream>>>(sproj, he_bf, convf, W_fe, W_ee,
                                                        ebuf, BT, T, H2);
        softmax_kernel<<<B, 256, 0, stream>>>(ebuf, lengths, alpha, T);
        gcalc_kernel<<<B, 128, 0, stream>>>(alpha, enc_bf, g, T, H2);

        // y = tanh(g@W_gy^T + b_gy + s2@W_sy^T) @ W_yy^T + b_yy  -> out[:, l, :]
        launch_gemm<false, false, false, false>(g,  H2, Wgy_bf, zbuf, hy, H, b_gy,
                                                B, H, H2, stream);
        launch_gemm<false, true,  true,  false>(s2, H,  Wsy_bf, zbuf, hy, H, nullptr,
                                                B, H, H, stream);
        launch_gemm<false, false, false, false>(hy, H,  Wyy_bf, zbuf,
                                                out + (size_t)l * V, L * V, b_yy,
                                                B, V, H, stream);

        // r1 = E_ys[tok] + s1@W_ss1^T + g@W_gs1^T + b_gs1; LSTM cell 1
        r1init_kernel<<<(B * H4 + 255) / 256, 256, 0, stream>>>(target, E_ys, b_gs1, r1,
                                                                l, L, H4, B * H4);
        launch_gemm<false, true, false, false>(s1, H,  Wss1_bf, zbuf, r1, H4, nullptr,
                                               B, H4, H,  stream);
        launch_gemm<false, true, false, false>(g,  H2, Wgs1_bf, zbuf, r1, H4, nullptr,
                                               B, H4, H2, stream);
        lstm_kernel<<<(B * H + 255) / 256, 256, 0, stream>>>(r1, s1, c1, H, B * H);

        // r2 = s1@W_ss12^T + s2@W_ss2^T + g@W_gs2^T + b_gs2; LSTM cell 2
        launch_gemm<false, false, false, false>(s1, H,  Wss12_bf, zbuf, r2, H4, b_gs2,
                                                B, H4, H, stream);
        launch_gemm<false, true,  false, false>(s2, H,  Wss2_bf,  zbuf, r2, H4, nullptr,
                                                B, H4, H,  stream);
        launch_gemm<false, true,  false, false>(g,  H2, Wgs2_bf,  zbuf, r2, H4, nullptr,
                                                B, H4, H2, stream);
        lstm_kernel<<<(B * H + 255) / 256, 256, 0, stream>>>(r2, s2, c2, H, B * H);
    }
}
